// CATANet_2018634629369
// MI455X (gfx1250) — compile-verified, matrix operand order checked by reading, not run on hardware
//
#include <hip/hip_runtime.h>
#include <hip/hip_bf16.h>

// ---------------- WMMA types & helpers (gfx1250, wave32) ----------------
typedef __attribute__((ext_vector_type(16))) _Float16 v16h;
typedef __attribute__((ext_vector_type(8)))  float    v8f;

__device__ __forceinline__ v8f wmma_f16(v16h a, v16h b, v8f c) {
  return __builtin_amdgcn_wmma_f32_16x16x32_f16(false, a, false, b, (short)0, c, false, false);
}

// A/B fragment (16x32 f16): lane = row (A) / col (B), kb mapping per ISA 16-bit layouts.
__device__ __forceinline__ v16h frag_ld_h(const _Float16* p, int ld) {
  int lane = threadIdx.x & 31;
  int r = lane & 15, kh = lane >> 4;
  v16h f;
#pragma unroll
  for (int v = 0; v < 8; ++v) {
    int kb = (v >> 2) * 16 + kh * 8 + (v & 3) * 2;
    f[2 * v]     = p[r * ld + kb];
    f[2 * v + 1] = p[r * ld + kb + 1];
  }
  return f;
}
// Logical K=16 (head dim), zero-padded to 32; row-major [16 x ld] halfs.
__device__ __forceinline__ v16h frag_ld_h16(const _Float16* p, int ld) {
  int lane = threadIdx.x & 31;
  int r = lane & 15, kh = lane >> 4;
  v16h f;
#pragma unroll
  for (int v = 0; v < 8; ++v) {
    int kb = (v >> 2) * 16 + kh * 8 + (v & 3) * 2;
    if (kb < 16) { f[2 * v] = p[r * ld + kb]; f[2 * v + 1] = p[r * ld + kb + 1]; }
    else         { f[2 * v] = (_Float16)0.f;  f[2 * v + 1] = (_Float16)0.f; }
  }
  return f;
}
// B fragment where element (k,n) lives at p[k*ld + n]  (V kept row-major [keys][dims]).
__device__ __forceinline__ v16h frag_ld_h_colT(const _Float16* p, int ld) {
  int lane = threadIdx.x & 31;
  int n = lane & 15, kh = lane >> 4;
  v16h f;
#pragma unroll
  for (int v = 0; v < 8; ++v) {
    int kb = (v >> 2) * 16 + kh * 8 + (v & 3) * 2;
    f[2 * v]     = p[kb * ld + n];
    f[2 * v + 1] = p[(kb + 1) * ld + n];
  }
  return f;
}
// From global fp32, full K=32, convert to f16 on load.
__device__ __forceinline__ v16h frag_ld_f32(const float* p, int ld) {
  int lane = threadIdx.x & 31;
  int r = lane & 15, kh = lane >> 4;
  v16h f;
#pragma unroll
  for (int v = 0; v < 8; ++v) {
    int kb = (v >> 2) * 16 + kh * 8 + (v & 3) * 2;
    f[2 * v]     = (_Float16)p[r * ld + kb];
    f[2 * v + 1] = (_Float16)p[r * ld + kb + 1];
  }
  return f;
}

// CDNA5 async global->LDS copy (16B per lane), tracked by ASYNCcnt.
__device__ __forceinline__ void async_copy16B(void* lds, const void* g) {
  uint32_t off = (uint32_t)(size_t)lds;   // low 32 bits of flat shared address = LDS offset
  asm volatile("global_load_async_to_lds_b128 %0, %1, off" :: "v"(off), "v"(g) : "memory");
}
__device__ __forceinline__ void async_wait0() {
  asm volatile("s_wait_asynccnt 0x0" ::: "memory");
}

__device__ __forceinline__ float hmax16(float v) {
#pragma unroll
  for (int o = 1; o < 16; o <<= 1) v = fmaxf(v, __shfl_xor(v, o, 32));
  return v;
}
__device__ __forceinline__ float hsum16(float v) {
#pragma unroll
  for (int o = 1; o < 16; o <<= 1) v += __shfl_xor(v, o, 32);
  return v;
}
__device__ __forceinline__ float gelu_exact(float x) {
  return 0.5f * x * (1.f + erff(x * 0.70710678118654752f));
}

// ---------------- fragment-resident softmax attention (one 16-row q tile) -------------
// K16: [Lk][16] halfs.  V16: [Lk][16] halfs (row-major, B-fragment gathers columns).
__device__ v8f attend16(v16h qf, const _Float16* K16, const _Float16* V16, int Lk,
                        _Float16* pscr, float scale) {
  int lane = threadIdx.x & 31;
  int n = lane & 15, gg = lane >> 4;
  v8f zero = {0.f, 0.f, 0.f, 0.f, 0.f, 0.f, 0.f, 0.f};
  float rmax[8];
#pragma unroll
  for (int r = 0; r < 8; ++r) rmax[r] = -1e30f;
  for (int t = 0; t < (Lk >> 4); ++t) {
    v8f s = wmma_f16(qf, frag_ld_h16(K16 + t * 16 * 16, 16), zero);
#pragma unroll
    for (int r = 0; r < 8; ++r) rmax[r] = fmaxf(rmax[r], s[r]);
  }
#pragma unroll
  for (int r = 0; r < 8; ++r) rmax[r] = hmax16(rmax[r]) * scale;
  float rsum[8];
#pragma unroll
  for (int r = 0; r < 8; ++r) rsum[r] = 0.f;
  v8f acc = zero;
  for (int c0 = 0; c0 < Lk; c0 += 32) {
    v8f s0 = wmma_f16(qf, frag_ld_h16(K16 + c0 * 16, 16), zero);
    v8f s1 = wmma_f16(qf, frag_ld_h16(K16 + (c0 + 16) * 16, 16), zero);
#pragma unroll
    for (int r = 0; r < 8; ++r) {
      float e0 = __expf(s0[r] * scale - rmax[r]);
      float e1 = __expf(s1[r] * scale - rmax[r]);
      rsum[r] += e0 + e1;
      pscr[(r + 8 * gg) * 32 + n]      = (_Float16)e0;
      pscr[(r + 8 * gg) * 32 + 16 + n] = (_Float16)e1;
    }
    asm volatile("s_wait_dscnt 0" ::: "memory");   // cross-lane LDS RAW before fragment reload
    v16h pf = frag_ld_h(pscr, 32);
    v16h vf = frag_ld_h_colT(V16 + (size_t)c0 * 16, 16);
    acc = wmma_f16(pf, vf, acc);
  }
#pragma unroll
  for (int r = 0; r < 8; ++r) rsum[r] = hsum16(rsum[r]);
#pragma unroll
  for (int r = 0; r < 8; ++r) acc[r] = acc[r] / rsum[r];
  return acc;
}

// ---------------- constants ----------------
#define HC 4
#define Nn 20736
#define Hh 144
#define Ww 144
#define Bb 2
#define BN 41472
#define NGc 162
#define NPp 121
#define BPp 242
#define BPT 61952

// ---------------- generic WMMA GEMM: Y[M,N] = X[M,K] @ W[N,K]^T (+bias)(+gelu) -------
template <typename TO>
__global__ __launch_bounds__(32) void gemm_wmma(const float* __restrict__ X,
                                                const float* __restrict__ Wt,
                                                const float* __restrict__ bias,
                                                TO* __restrict__ Y,
                                                int K, int Nc, int act) {
  int tm = blockIdx.x, tn = blockIdx.y;
  int lane = threadIdx.x;
  v8f acc = {0.f, 0.f, 0.f, 0.f, 0.f, 0.f, 0.f, 0.f};
  const float* xr = X + (size_t)tm * 16 * K;
  const float* wr = Wt + (size_t)tn * 16 * K;
  for (int k0 = 0; k0 < K; k0 += 32) {
    acc = wmma_f16(frag_ld_f32(xr + k0, K), frag_ld_f32(wr + k0, K), acc);
  }
  int n = lane & 15, gg = lane >> 4;
  int col = tn * 16 + n;
  float bv = bias ? bias[col] : 0.f;
#pragma unroll
  for (int r = 0; r < 8; ++r) {
    float v = acc[r] + bv;
    if (act == 1) v = gelu_exact(v);
    Y[((size_t)(tm * 16 + r + 8 * gg)) * Nc + col] = (TO)v;
  }
}

// ---------------- WMMA implicit-GEMM convs ----------------
// head: K=27 (3ch*3x3) padded to 32, tokens x 64oc
__global__ __launch_bounds__(32) void head_conv_wmma(const float* __restrict__ img,
                                                     const float* __restrict__ w,
                                                     const float* __restrict__ bsc,
                                                     float* __restrict__ x0) {
  int tm = blockIdx.x, tn = blockIdx.y;
  int lane = threadIdx.x, r = lane & 15, kh = lane >> 4;
  int token = tm * 16 + r;
  int b = token / Nn, pos = token % Nn;
  int hh = pos / Ww, ww = pos % Ww;
  v16h af, bf;
#pragma unroll
  for (int v = 0; v < 8; ++v) {
    int kb = (v >> 2) * 16 + kh * 8 + (v & 3) * 2;
#pragma unroll
    for (int j = 0; j < 2; ++j) {
      int k = kb + j;
      float a = 0.f, bw = 0.f;
      if (k < 27) {
        int ic = k / 9, q = k - ic * 9, ky = q / 3, kx = q - ky * 3;
        int y = hh + ky - 1, x = ww + kx - 1;
        if (y >= 0 && y < Hh && x >= 0 && x < Ww)
          a = img[(((size_t)b * 3 + ic) * Hh + y) * Ww + x];
        bw = w[(tn * 16 + r) * 27 + k];
      }
      af[2 * v + j] = (_Float16)a;
      bf[2 * v + j] = (_Float16)bw;
    }
  }
  v8f zero = {0.f, 0.f, 0.f, 0.f, 0.f, 0.f, 0.f, 0.f};
  v8f acc = wmma_f16(af, bf, zero);
  int n = lane & 15, gg = lane >> 4;
  int col = tn * 16 + n;
#pragma unroll
  for (int rr = 0; rr < 8; ++rr)
    x0[((size_t)(tm * 16 + rr + 8 * gg)) * 64 + col] = acc[rr] + bsc[col];
}

// 3x3 conv 64->64, + bias + feat residual; K = 576 (ic*9 + ky*3 + kx, weight-contiguous)
__global__ __launch_bounds__(32) void tail_conv_wmma(const float* __restrict__ x,
                                                     const float* __restrict__ w,
                                                     const float* __restrict__ bsc,
                                                     const float* __restrict__ feat,
                                                     float* __restrict__ out) {
  int tm = blockIdx.x, tn = blockIdx.y;
  int lane = threadIdx.x, r = lane & 15, kh = lane >> 4;
  int token = tm * 16 + r;
  int b = token / Nn, pos = token % Nn;
  int hh = pos / Ww, ww = pos % Ww;
  const float* xb = x + (size_t)b * Nn * 64;
  v8f acc = {0.f, 0.f, 0.f, 0.f, 0.f, 0.f, 0.f, 0.f};
  for (int k0 = 0; k0 < 576; k0 += 32) {
    v16h af, bf;
#pragma unroll
    for (int v = 0; v < 8; ++v) {
      int kb = (v >> 2) * 16 + kh * 8 + (v & 3) * 2 + k0;
#pragma unroll
      for (int j = 0; j < 2; ++j) {
        int k = kb + j;
        int ic = k / 9, q = k - ic * 9, ky = q / 3, kx = q - ky * 3;
        int y = hh + ky - 1, xx = ww + kx - 1;
        float a = (y >= 0 && y < Hh && xx >= 0 && xx < Ww)
                    ? xb[((size_t)y * Ww + xx) * 64 + ic] : 0.f;
        af[2 * v + j] = (_Float16)a;
        bf[2 * v + j] = (_Float16)w[(tn * 16 + r) * 576 + k];
      }
    }
    acc = wmma_f16(af, bf, acc);
  }
  int n = lane & 15, gg = lane >> 4;
  int col = tn * 16 + n;
#pragma unroll
  for (int rr = 0; rr < 8; ++rr) {
    size_t o = ((size_t)(tm * 16 + rr + 8 * gg)) * 64 + col;
    out[o] = acc[rr] + bsc[col] + feat[o];
  }
}

// 3x3 conv 64->12 + pixel shuffle (x2); N padded 12->16
__global__ __launch_bounds__(32) void up_conv_wmma(const float* __restrict__ body,
                                                   const float* __restrict__ w,
                                                   const float* __restrict__ bsc,
                                                   float* __restrict__ out) {
  int tm = blockIdx.x;
  int lane = threadIdx.x, r = lane & 15, kh = lane >> 4;
  int token = tm * 16 + r;
  int b = token / Nn, pos = token % Nn;
  int hh = pos / Ww, ww = pos % Ww;
  const float* xb = body + (size_t)b * Nn * 64;
  v8f acc = {0.f, 0.f, 0.f, 0.f, 0.f, 0.f, 0.f, 0.f};
  for (int k0 = 0; k0 < 576; k0 += 32) {
    v16h af, bf;
#pragma unroll
    for (int v = 0; v < 8; ++v) {
      int kb = (v >> 2) * 16 + kh * 8 + (v & 3) * 2 + k0;
#pragma unroll
      for (int j = 0; j < 2; ++j) {
        int k = kb + j;
        int ic = k / 9, q = k - ic * 9, ky = q / 3, kx = q - ky * 3;
        int y = hh + ky - 1, xx = ww + kx - 1;
        float a = (y >= 0 && y < Hh && xx >= 0 && xx < Ww)
                    ? xb[((size_t)y * Ww + xx) * 64 + ic] : 0.f;
        af[2 * v + j] = (_Float16)a;
        bf[2 * v + j] = (_Float16)((r < 12) ? w[r * 576 + k] : 0.f);
      }
    }
    acc = wmma_f16(af, bf, acc);
  }
  int cc = lane & 15, gg = lane >> 4;
  if (cc < 12) {
    int oc = cc >> 2, r1 = (cc >> 1) & 1, r2 = cc & 1;
#pragma unroll
    for (int rr = 0; rr < 8; ++rr) {
      int t2 = tm * 16 + rr + 8 * gg;
      int b2 = t2 / Nn, p2 = t2 % Nn;
      int h2 = p2 / Ww, w2 = p2 % Ww;
      out[(((size_t)b2 * 3 + oc) * 288 + (h2 * 2 + r1)) * 288 + (w2 * 2 + r2)] = acc[rr] + bsc[cc];
    }
  }
}

// ---------------- IASA attention: block = (group, head, batch); async-staged K/V -------
__global__ __launch_bounds__(256) void iasa_attn(const _Float16* __restrict__ qs,
                                                 const _Float16* __restrict__ ks,
                                                 const _Float16* __restrict__ vs,
                                                 const _Float16* __restrict__ kg,
                                                 const _Float16* __restrict__ vg,
                                                 float* __restrict__ out) {
  int g = blockIdx.x, h = blockIdx.y, b = blockIdx.z;
  __shared__ _Float16 k_sh[256][16];
  __shared__ _Float16 v_sh[256][16];
  __shared__ _Float16 kg_sh[64][16];
  __shared__ _Float16 vg_sh[64][16];
  __shared__ _Float16 p_sh[8][512];
  int tid = threadIdx.x;
  const _Float16* kbp = ks + ((size_t)b * Nn) * 64 + h * 16;
  const _Float16* vbp = vs + ((size_t)b * Nn) * 64 + h * 16;
  // 16-byte async chunks: 512 K + 512 V + 128 kg + 128 vg = 1280 == 5*256
  for (int j = tid; j < 1280; j += 256) {
    const _Float16* gsrc;
    _Float16* ldst;
    if (j < 1024) {
      int jj = j & 511;
      int row = jj >> 1, hf = jj & 1;
      int pp = g * 128 + row;
      if (pp >= Nn) pp = 2 * Nn - 1 - pp;      // reflected window tail
      const _Float16* base = (j < 512) ? kbp : vbp;
      gsrc = base + (size_t)pp * 64 + hf * 8;
      ldst = (j < 512) ? &k_sh[row][hf * 8] : &v_sh[row][hf * 8];
    } else {
      int jj = j - 1024;
      int row = (jj & 127) >> 1, hf = jj & 1;
      const _Float16* base = (jj < 128) ? kg : vg;
      gsrc = base + row * 64 + h * 16 + hf * 8;
      ldst = (jj < 128) ? &kg_sh[row][hf * 8] : &vg_sh[row][hf * 8];
    }
    async_copy16B(ldst, gsrc);
  }
  async_wait0();
  __syncthreads();
  int wave = tid >> 5, lane = tid & 31;
  v16h qf = frag_ld_h16(qs + ((size_t)(b * Nn + g * 128 + wave * 16)) * 64 + h * 16, 64);
  v8f o1 = attend16(qf, &k_sh[0][0], &v_sh[0][0], 256, p_sh[wave], 0.25f);
  v8f o2 = attend16(qf, &kg_sh[0][0], &vg_sh[0][0], 64, p_sh[wave], 0.25f);
  float* ob = out + ((size_t)b * Nn) * 64 + h * 16;
  int n = lane & 15, gg = lane >> 4;
#pragma unroll
  for (int r = 0; r < 8; ++r) {
    int row = g * 128 + wave * 16 + r + 8 * gg;
    ob[(size_t)row * 64 + n] = o1[r] + o2[r];
  }
}

// ---------------- LRSA patch attention: block = (batch*patch, head); async-staged -----
__global__ __launch_bounds__(256) void lrsa_attn_k(const _Float16* __restrict__ q,
                                                   const _Float16* __restrict__ k,
                                                   const _Float16* __restrict__ v,
                                                   float* __restrict__ out) {
  int p = blockIdx.x, h = blockIdx.y;
  __shared__ _Float16 k_sh[256][16];
  __shared__ _Float16 v_sh[256][16];
  __shared__ _Float16 p_sh[8][512];
  int tid = threadIdx.x;
  const _Float16* kb = k + ((size_t)p * 256) * 64 + h * 16;
  const _Float16* vb = v + ((size_t)p * 256) * 64 + h * 16;
  for (int j = tid; j < 1024; j += 256) {
    int jj = j & 511;
    int row = jj >> 1, hf = jj & 1;
    const _Float16* gsrc = ((j < 512) ? kb : vb) + (size_t)row * 64 + hf * 8;
    _Float16* ldst = (j < 512) ? &k_sh[row][hf * 8] : &v_sh[row][hf * 8];
    async_copy16B(ldst, gsrc);
  }
  async_wait0();
  __syncthreads();
  int wave = tid >> 5, lane = tid & 31;
  int n = lane & 15, gg = lane >> 4;
  for (int qt = wave; qt < 16; qt += 8) {
    v16h qf = frag_ld_h16(q + ((size_t)(p * 256 + qt * 16)) * 64 + h * 16, 64);
    v8f o = attend16(qf, &k_sh[0][0], &v_sh[0][0], 256, p_sh[wave], 0.25f);
    float* ob = out + ((size_t)(p * 256 + qt * 16)) * 64 + h * 16;
#pragma unroll
    for (int r = 0; r < 8; ++r) ob[(size_t)(r + 8 * gg) * 64 + n] = o[r];
  }
}

// ---------------- small / elementwise kernels ----------------
__global__ __launch_bounds__(256) void layernorm64(const float* __restrict__ in,
                                                   const float* __restrict__ g,
                                                   const float* __restrict__ b,
                                                   float* __restrict__ out, int rows) {
  int wave = threadIdx.x >> 5, lane = threadIdx.x & 31;
  int row = blockIdx.x * 8 + wave;
  if (row >= rows) return;
  const float* p = in + (size_t)row * 64;
  float v0 = p[lane], v1 = p[lane + 32];
  float s = v0 + v1;
#pragma unroll
  for (int o = 1; o < 32; o <<= 1) s += __shfl_xor(s, o, 32);
  float mu = s * (1.f / 64.f);
  float d0 = v0 - mu, d1 = v1 - mu;
  float q = d0 * d0 + d1 * d1;
#pragma unroll
  for (int o = 1; o < 32; o <<= 1) q += __shfl_xor(q, o, 32);
  float inv = rsqrtf(q * (1.f / 64.f) + 1e-5f);
  float* po = out + (size_t)row * 64;
  po[lane] = d0 * inv * g[lane] + b[lane];
  po[lane + 32] = d1 * inv * g[lane + 32] + b[lane + 32];
}

__global__ void normalize_rows64(const float* __restrict__ in, float* __restrict__ out) {
  int r = threadIdx.x;
  float s = 0.f;
  for (int k = 0; k < 64; ++k) { float v = in[r * 64 + k]; s += v * v; }
  float inv = 1.f / fmaxf(sqrtf(s), 1e-12f);
  for (int k = 0; k < 64; ++k) out[r * 64 + k] = in[r * 64 + k] * inv;
}

__global__ __launch_bounds__(256) void row_argmax64(const float* __restrict__ s,
                                                    int* __restrict__ out, int rows) {
  int wave = threadIdx.x >> 5, lane = threadIdx.x & 31;
  int row = blockIdx.x * 8 + wave;
  if (row >= rows) return;
  const float* p = s + (size_t)row * 64;
  float bv = p[lane]; int bi = lane;
  float v1 = p[lane + 32];
  if (v1 > bv) { bv = v1; bi = lane + 32; }
#pragma unroll
  for (int o = 1; o < 32; o <<= 1) {
    float ov = __shfl_xor(bv, o, 32);
    int oi = __shfl_xor(bi, o, 32);
    if (ov > bv || (ov == bv && oi < bi)) { bv = ov; bi = oi; }
  }
  if (lane == 0) out[row] = bi;
}

__global__ __launch_bounds__(128) void bucket_hist(const int* __restrict__ buckets,
                                                   int* __restrict__ chunkcnt) {
  int b = blockIdx.y, ch = blockIdx.x;
  __shared__ int cnt[64];
  if (threadIdx.x < 64) cnt[threadIdx.x] = 0;
  __syncthreads();
  atomicAdd(&cnt[buckets[b * Nn + ch * 128 + threadIdx.x]], 1);
  __syncthreads();
  if (threadIdx.x < 64) chunkcnt[(b * NGc + ch) * 64 + threadIdx.x] = cnt[threadIdx.x];
}

__global__ __launch_bounds__(64) void bucket_scan(const int* __restrict__ chunkcnt,
                                                  int* __restrict__ chunkbase) {
  int b = blockIdx.x, c = threadIdx.x;
  __shared__ int tot[64];
  __shared__ int base[64];
  int s = 0;
  for (int ch = 0; ch < NGc; ++ch) s += chunkcnt[(b * NGc + ch) * 64 + c];
  tot[c] = s;
  __syncthreads();
  if (c == 0) { int run = 0; for (int k = 0; k < 64; ++k) { base[k] = run; run += tot[k]; } }
  __syncthreads();
  int run = base[c];
  for (int ch = 0; ch < NGc; ++ch) {
    chunkbase[(b * NGc + ch) * 64 + c] = run;
    run += chunkcnt[(b * NGc + ch) * 64 + c];
  }
}

__global__ __launch_bounds__(128) void bucket_scatter(const int* __restrict__ buckets,
                                                      const int* __restrict__ chunkbase,
                                                      int* __restrict__ idx) {
  int b = blockIdx.y, ch = blockIdx.x;
  __shared__ int loc[128];
  int t = threadIdx.x;
  int e = ch * 128 + t;
  int myb = buckets[b * Nn + e];
  loc[t] = myb;
  __syncthreads();
  int rank = 0;
  for (int j = 0; j < t; ++j) rank += (loc[j] == myb);
  idx[b * Nn + chunkbase[(b * NGc + ch) * 64 + myb] + rank] = e;   // stable argsort
}

__global__ __launch_bounds__(256) void gather_rows64h(const _Float16* __restrict__ src,
                                                      const int* __restrict__ idx,
                                                      _Float16* __restrict__ dst) {
  size_t i = (size_t)blockIdx.x * 256 + threadIdx.x;
  if (i >= (size_t)BN * 64) return;
  int c = i & 63; size_t r = i >> 6;
  int b = (int)(r / Nn), j = (int)(r % Nn);
  dst[i] = src[(((size_t)b * Nn + idx[(size_t)b * Nn + j]) << 6) + c];
}

__global__ __launch_bounds__(256) void scatter_rows64(const float* __restrict__ src,
                                                      const int* __restrict__ idx,
                                                      float* __restrict__ dst) {
  size_t i = (size_t)blockIdx.x * 256 + threadIdx.x;
  if (i >= (size_t)BN * 64) return;
  int c = i & 63; size_t r = i >> 6;
  int b = (int)(r / Nn), j = (int)(r % Nn);
  dst[(((size_t)b * Nn + idx[(size_t)b * Nn + j]) << 6) + c] = src[i];
}

__global__ __launch_bounds__(256) void add_vec(const float* __restrict__ a,
                                               const float* __restrict__ b,
                                               float* __restrict__ o, size_t n) {
  size_t i = (size_t)blockIdx.x * 256 + threadIdx.x;
  if (i < n) o[i] = a[i] + b[i];
}

__global__ __launch_bounds__(256) void zero_vec(float* __restrict__ p, size_t n) {
  size_t i = (size_t)blockIdx.x * 256 + threadIdx.x;
  if (i < n) p[i] = 0.f;
}

__global__ __launch_bounds__(256) void dwconv5_gelu_add(const float* __restrict__ in,
                                                        const float* __restrict__ w,
                                                        const float* __restrict__ bsc,
                                                        float* __restrict__ out) {
  size_t i = (size_t)blockIdx.x * 256 + threadIdx.x;
  if (i >= (size_t)BN * 128) return;
  int c = (int)(i & 127); size_t r = i >> 7;
  int pos = (int)(r % Nn); int b = (int)(r / Nn);
  int hh = pos / Ww, ww = pos % Ww;
  float s = bsc[c];
  for (int ky = 0; ky < 5; ++ky) {
    int y = hh + ky - 2; if (y < 0 || y >= Hh) continue;
    for (int kx = 0; kx < 5; ++kx) {
      int x = ww + kx - 2; if (x < 0 || x >= Ww) continue;
      s += w[(c * 5 + ky) * 5 + kx] * in[(((size_t)b * Nn) + (size_t)y * Ww + x) * 128 + c];
    }
  }
  out[i] = in[i] + gelu_exact(s);
}

__global__ __launch_bounds__(256) void lrsa_crop(const float* __restrict__ x, float* __restrict__ t) {
  size_t i = (size_t)blockIdx.x * 256 + threadIdx.x;
  if (i >= (size_t)BPT * 64) return;
  int c = (int)(i & 63); size_t r = i >> 6;
  int tok = (int)(r & 255); int bp = (int)(r >> 8);
  int b = bp / NPp, p = bp % NPp;
  int top = (p / 11) * 14;  if (top > 128) top = 128;
  int left = (p % 11) * 14; if (left > 128) left = 128;
  int pos = (top + (tok >> 4)) * Ww + left + (tok & 15);
  t[i] = x[(((size_t)b * Nn + pos) << 6) + c];
}

__global__ __launch_bounds__(256) void lrsa_scatter_add(const float* __restrict__ t,
                                                        float* __restrict__ acc) {
  size_t i = (size_t)blockIdx.x * 256 + threadIdx.x;
  if (i >= (size_t)BPT * 64) return;
  int c = (int)(i & 63); size_t r = i >> 6;
  int tok = (int)(r & 255); int bp = (int)(r >> 8);
  int b = bp / NPp, p = bp % NPp;
  int top = (p / 11) * 14;  if (top > 128) top = 128;
  int left = (p % 11) * 14; if (left > 128) left = 128;
  int pos = (top + (tok >> 4)) * Ww + left + (tok & 15);
  atomicAdd(&acc[(((size_t)b * Nn + pos) << 6) + c], t[i]);
}

__global__ __launch_bounds__(256) void overlap_weight(float* __restrict__ acc) {
  size_t i = (size_t)blockIdx.x * 256 + threadIdx.x;
  if (i >= (size_t)BN * 64) return;
  int pos = (int)((i >> 6) % Nn);
  int hh = pos / Ww, ww = pos % Ww;
  float wgt = 1.f;
  for (int k = 14; k < 142; k += 14) {
    int top = (k + 16 > 144) ? 128 : k;
    int down = k + 2;
    if (hh >= top && hh < down) wgt *= 0.5f;
    if (ww >= top && ww < down) wgt *= 0.5f;
  }
  acc[i] *= wgt;
}

// ---------------- host orchestration ----------------
extern "C" void kernel_launch(void* const* d_in, const int* in_sizes, int n_in,
                              void* d_out, int out_size, void* d_ws, size_t ws_size,
                              hipStream_t stream) {
  const float* img      = (const float*)d_in[0];
  const float* head_w   = (const float*)d_in[1];
  const float* head_b   = (const float*)d_in[2];
  const float* ln1_g    = (const float*)d_in[3];
  const float* ln1_b    = (const float*)d_in[4];
  const float* means    = (const float*)d_in[5];
  const float* irca_k_w = (const float*)d_in[6];
  const float* irca_v_w = (const float*)d_in[7];
  const float* iasa_q_w = (const float*)d_in[8];
  const float* iasa_k_w = (const float*)d_in[9];
  const float* iasa_v_w = (const float*)d_in[10];
  const float* iasa_p_w = (const float*)d_in[11];
  const float* ln2_g    = (const float*)d_in[12];
  const float* ln2_b    = (const float*)d_in[13];
  const float* fc1_w    = (const float*)d_in[14];
  const float* fc1_b    = (const float*)d_in[15];
  const float* dw_w     = (const float*)d_in[16];
  const float* dw_b     = (const float*)d_in[17];
  const float* fc2_w    = (const float*)d_in[18];
  const float* fc2_b    = (const float*)d_in[19];
  const float* lln_a_g  = (const float*)d_in[20];
  const float* lln_a_b  = (const float*)d_in[21];
  const float* lq_w     = (const float*)d_in[22];
  const float* lk_w     = (const float*)d_in[23];
  const float* lv_w     = (const float*)d_in[24];
  const float* lp_w     = (const float*)d_in[25];
  const float* lln_f_g  = (const float*)d_in[26];
  const float* lln_f_b  = (const float*)d_in[27];
  const float* lfc1_w   = (const float*)d_in[28];
  const float* lfc1_b   = (const float*)d_in[29];
  const float* ldw_w    = (const float*)d_in[30];
  const float* ldw_b    = (const float*)d_in[31];
  const float* lfc2_w   = (const float*)d_in[32];
  const float* lfc2_b   = (const float*)d_in[33];
  const float* tail_w   = (const float*)d_in[34];
  const float* tail_b   = (const float*)d_in[35];
  const float* up_w     = (const float*)d_in[36];
  const float* up_b     = (const float*)d_in[37];
  (void)n_in; (void)in_sizes; (void)out_size; (void)ws_size;

  char* base = (char*)d_ws;
  size_t off = 0;
  auto alloc = [&](size_t nfloats) { float* p = (float*)(base + off); off += nfloats * sizeof(float); return p; };
  const size_t SL = (size_t)BN * 128;
  float* X0 = alloc((size_t)BN * 64);
  float* S1 = alloc(SL); float* S2 = alloc(SL); float* S3 = alloc(SL); float* S4 = alloc(SL);
  float* S5 = alloc(SL); float* S6 = alloc(SL); float* S7 = alloc(SL); float* S8 = alloc(SL);
  float* MH = alloc(64 * 64);
  _Float16* KGH = (_Float16*)alloc(2048);
  _Float16* VGH = (_Float16*)alloc(2048);
  int* IB    = (int*)alloc(BN);
  int* IDX   = (int*)alloc(BN);
  int* CCNT  = (int*)alloc((size_t)Bb * NGc * 64);
  int* CBASE = (int*)alloc((size_t)Bb * NGc * 64);

  const int TBN64  = (BN * 64 + 255) / 256;
  const int TBN128 = (int)(((size_t)BN * 128 + 255) / 256);
  const int TBPT64 = (int)(((size_t)BPT * 64 + 255) / 256);

  // ---- head conv (WMMA implicit GEMM) -> tokens
  head_conv_wmma<<<dim3(BN / 16, 4), 32, 0, stream>>>(img, head_w, head_b, X0);
  // ---- LN1, WMMA bucket scoring, stable counting argsort
  layernorm64<<<BN / 8, 256, 0, stream>>>(X0, ln1_g, ln1_b, S1, BN);
  normalize_rows64<<<1, 64, 0, stream>>>(means, MH);
  gemm_wmma<float><<<dim3(BN / 16, 4), 32, 0, stream>>>(S1, MH, nullptr, S8, 64, 64, 0);
  row_argmax64<<<BN / 8, 256, 0, stream>>>(S8, IB, BN);
  bucket_hist<<<dim3(NGc, Bb), 128, 0, stream>>>(IB, CCNT);
  bucket_scan<<<Bb, 64, 0, stream>>>(CCNT, CBASE);
  bucket_scatter<<<dim3(NGc, Bb), 128, 0, stream>>>(IB, CBASE, IDX);
  // ---- global tokens (f16 out)
  gemm_wmma<_Float16><<<dim3(4, 4), 32, 0, stream>>>(means, irca_k_w, nullptr, KGH, 64, 64, 0);
  gemm_wmma<_Float16><<<dim3(4, 4), 32, 0, stream>>>(means, irca_v_w, nullptr, VGH, 64, 64, 0);
  // ---- q/k/v projections straight to f16, gather into sorted order
  gemm_wmma<_Float16><<<dim3(BN / 16, 4), 32, 0, stream>>>(S1, iasa_q_w, nullptr, (_Float16*)S2, 64, 64, 0);
  gemm_wmma<_Float16><<<dim3(BN / 16, 4), 32, 0, stream>>>(S1, iasa_k_w, nullptr, (_Float16*)S3, 64, 64, 0);
  gemm_wmma<_Float16><<<dim3(BN / 16, 4), 32, 0, stream>>>(S1, iasa_v_w, nullptr, (_Float16*)S4, 64, 64, 0);
  gather_rows64h<<<TBN64, 256, 0, stream>>>((_Float16*)S2, IDX, (_Float16*)S5);
  gather_rows64h<<<TBN64, 256, 0, stream>>>((_Float16*)S3, IDX, (_Float16*)S6);
  gather_rows64h<<<TBN64, 256, 0, stream>>>((_Float16*)S4, IDX, (_Float16*)S7);
  // ---- IASA attention (async-staged LDS, WMMA softmax)
  iasa_attn<<<dim3(NGc, HC, Bb), 256, 0, stream>>>((_Float16*)S5, (_Float16*)S6, (_Float16*)S7,
                                                   KGH, VGH, S8);
  scatter_rows64<<<TBN64, 256, 0, stream>>>(S8, IDX, S2);
  gemm_wmma<float><<<dim3(BN / 16, 4), 32, 0, stream>>>(S2, iasa_p_w, nullptr, S3, 64, 64, 0);
  add_vec<<<TBN64, 256, 0, stream>>>(S3, X0, S4, (size_t)BN * 64);          // x1
  // ---- conv FFN
  layernorm64<<<BN / 8, 256, 0, stream>>>(S4, ln2_g, ln2_b, S1, BN);
  gemm_wmma<float><<<dim3(BN / 16, 8), 32, 0, stream>>>(S1, fc1_w, fc1_b, S5, 64, 128, 1);
  dwconv5_gelu_add<<<TBN128, 256, 0, stream>>>(S5, dw_w, dw_b, S6);
  gemm_wmma<float><<<dim3(BN / 16, 4), 32, 0, stream>>>(S6, fc2_w, fc2_b, S7, 128, 64, 0);
  add_vec<<<TBN64, 256, 0, stream>>>(S7, S4, S2, (size_t)BN * 64);          // x2
  // ---- LRSA
  lrsa_crop<<<TBPT64, 256, 0, stream>>>(S2, S5);
  layernorm64<<<BPT / 8, 256, 0, stream>>>(S5, lln_a_g, lln_a_b, S6, BPT);
  gemm_wmma<_Float16><<<dim3(BPT / 16, 4), 32, 0, stream>>>(S6, lq_w, nullptr, (_Float16*)S7, 64, 64, 0);
  gemm_wmma<_Float16><<<dim3(BPT / 16, 4), 32, 0, stream>>>(S6, lk_w, nullptr, (_Float16*)S8, 64, 64, 0);
  gemm_wmma<_Float16><<<dim3(BPT / 16, 4), 32, 0, stream>>>(S6, lv_w, nullptr, (_Float16*)S1, 64, 64, 0);
  lrsa_attn_k<<<dim3(BPp, HC), 256, 0, stream>>>((_Float16*)S7, (_Float16*)S8, (_Float16*)S1, S3);
  gemm_wmma<float><<<dim3(BPT / 16, 4), 32, 0, stream>>>(S3, lp_w, nullptr, S6, 64, 64, 0);
  add_vec<<<TBPT64, 256, 0, stream>>>(S6, S5, S6, (size_t)BPT * 64);        // t = attn + t
  // ---- overlap-add with 0.5 seam weighting
  zero_vec<<<TBN64, 256, 0, stream>>>(S4, (size_t)BN * 64);
  lrsa_scatter_add<<<TBPT64, 256, 0, stream>>>(S6, S4);
  overlap_weight<<<TBN64, 256, 0, stream>>>(S4);
  // ---- LRSA conv FFN
  layernorm64<<<BN / 8, 256, 0, stream>>>(S4, lln_f_g, lln_f_b, S1, BN);
  gemm_wmma<float><<<dim3(BN / 16, 8), 32, 0, stream>>>(S1, lfc1_w, lfc1_b, S7, 64, 128, 1);
  dwconv5_gelu_add<<<TBN128, 256, 0, stream>>>(S7, ldw_w, ldw_b, S8);
  gemm_wmma<float><<<dim3(BN / 16, 4), 32, 0, stream>>>(S8, lfc2_w, lfc2_b, S3, 128, 64, 0);
  add_vec<<<TBN64, 256, 0, stream>>>(S3, S4, S5, (size_t)BN * 64);          // x3
  // ---- tail conv + feat residual (WMMA), upsample + pixel shuffle (WMMA)
  tail_conv_wmma<<<dim3(BN / 16, 4), 32, 0, stream>>>(S5, tail_w, tail_b, X0, S6);
  up_conv_wmma<<<dim3(BN / 16, 1), 32, 0, stream>>>(S6, up_w, up_b, (float*)d_out);
}
